// GRU_61400852463623
// MI455X (gfx1250) — matrix operand order repaired and checked by reading, not compiled, not run on hardware
//
#include <hip/hip_runtime.h>
#include <hip/hip_bf16.h>

// GRU on MI455X (gfx1250, wave32, WMMA).
//   Phase 1: gi = x @ Wi + bi    -- parallel bf16 WMMA GEMM (one 16x16 tile per wave)
//   Phase 2: sequential scan     -- persistent 16-workgroup kernel, Wh prepacked in LDS
//            as bf16 WMMA B-fragments, global atomic barrier between time steps,
//            gi loads software-pipelined one step ahead (in flight across the barrier).

typedef __attribute__((ext_vector_type(16))) __bf16 v16bf;
typedef __attribute__((ext_vector_type(8)))  float  v8f;

#define GRU_B 64
#define GRU_T 1024
#define GRU_D 256
#define GRU_H 256
#define GRU_3H 768
#define NBLK 16          // persistent scan workgroups (one per 16-col tile of H)

__device__ __forceinline__ v8f wmma_bf16(v16bf a, v16bf b, v8f c) {
  return __builtin_amdgcn_wmma_f32_16x16x32_bf16(false, a, false, b, (short)0, c,
                                                 false, false);
}

// A-fragment (16x32 bf16, ISA layout): per lane, elements 0..7 = 8 contiguous
// floats at base, elements 8..15 = 8 contiguous floats at base+16.
// base must already include row*ld + kk*32 + (lane>>4)*8.
__device__ __forceinline__ v16bf load_a_frag(const float* __restrict__ base) {
  const float4* p = reinterpret_cast<const float4*>(base);
  float4 q0 = p[0], q1 = p[1], q2 = p[4], q3 = p[5];
  v16bf a;
  a[0]  = (__bf16)q0.x; a[1]  = (__bf16)q0.y; a[2]  = (__bf16)q0.z; a[3]  = (__bf16)q0.w;
  a[4]  = (__bf16)q1.x; a[5]  = (__bf16)q1.y; a[6]  = (__bf16)q1.z; a[7]  = (__bf16)q1.w;
  a[8]  = (__bf16)q2.x; a[9]  = (__bf16)q2.y; a[10] = (__bf16)q2.z; a[11] = (__bf16)q2.w;
  a[12] = (__bf16)q3.x; a[13] = (__bf16)q3.y; a[14] = (__bf16)q3.z; a[15] = (__bf16)q3.w;
  return a;
}

// B-fragment (32x16 bf16, ISA layout): lane L -> col = colBase + (L&15),
// K = k0 + (L>>4)*8 + e (e<8), k0 + 16 + (L>>4)*8 + (e-8) (e>=8). Gathered from row-major W.
__device__ __forceinline__ v16bf load_b_frag(const float* __restrict__ W, int ld,
                                             int k0, int colBase, int lane) {
  int l16 = lane & 15, lh = lane >> 4;
  int kb = k0 + lh * 8;
  const float* p = W + (size_t)kb * ld + colBase + l16;
  v16bf b;
#pragma unroll
  for (int e = 0; e < 16; ++e) b[e] = (__bf16)p[(size_t)(e < 8 ? e : e + 8) * ld];
  return b;
}

__device__ __forceinline__ float sigm(float v) { return 1.f / (1.f + __expf(-v)); }

// ---------------------------------------------------------------------------
// Phase 1: gi[b, t0+tl, :] = x[b, t0+tl, :] @ Wi + bi   for tl in [0, Tc)
// gi chunk layout: [B][Tc][768] row-major. One 16x16 output tile per wave.
// ---------------------------------------------------------------------------
__global__ void __launch_bounds__(256) gi_gemm_kernel(
    const float* __restrict__ x, const float* __restrict__ Wi,
    const float* __restrict__ bi, float* __restrict__ gi,
    int Tc, int tcsh, int t0) {
  int wv = threadIdx.x >> 5;
  int lane = threadIdx.x & 31, l16 = lane & 15, lh = lane >> 4;
  int wid = blockIdx.x * 8 + wv;
  int rt = wid / 48, ct = wid % 48;
  int colBase = ct * 16;

  // A source row: chunk row -> (b, tl) -> x row b*T + t0 + tl
  int rA = rt * 16 + l16;
  int bA = rA >> tcsh, tlA = rA & (Tc - 1);
  const float* xrow = x + ((size_t)bA * GRU_T + t0 + tlA) * GRU_D + lh * 8;

  v8f acc = {};
#pragma unroll
  for (int kk = 0; kk < 8; ++kk) {
    v16bf a = load_a_frag(xrow + kk * 32);
    v16bf b = load_b_frag(Wi, GRU_3H, kk * 32, colBase, lane);
    acc = wmma_bf16(a, b, acc);
  }

  int col = colBase + l16;
  float bc = bi[col];
#pragma unroll
  for (int v = 0; v < 8; ++v) {
    int r = rt * 16 + lh * 8 + v;  // chunk row (C/D layout: row = v + 8*lh)
    gi[(size_t)r * GRU_3H + col] = acc[v] + bc;
  }
}

// ---------------------------------------------------------------------------
// Grid barrier across the 16 persistent workgroups (sense-free, monotonic).
// cnt accumulates arrivals; the phase-th barrier completes when cnt == 16*phase.
// All threads acquire-spin on gen (no trailing __syncthreads on the wake path).
// ---------------------------------------------------------------------------
__device__ __forceinline__ void grid_barrier(unsigned* cnt, unsigned* gen,
                                             unsigned phase) {
  __syncthreads();  // all of this WG's stores for the step are issued
  if (threadIdx.x == 0) {
    __threadfence();  // make this WG's h'/ys stores device-visible
    unsigned a = __hip_atomic_fetch_add(cnt, 1u, __ATOMIC_ACQ_REL,
                                        __HIP_MEMORY_SCOPE_AGENT);
    if (a + 1u == phase * (unsigned)NBLK)
      __hip_atomic_fetch_add(gen, 1u, __ATOMIC_ACQ_REL, __HIP_MEMORY_SCOPE_AGENT);
  }
  while (__hip_atomic_load(gen, __ATOMIC_ACQUIRE, __HIP_MEMORY_SCOPE_AGENT) < phase)
    __builtin_amdgcn_s_sleep(1);
}

// ---------------------------------------------------------------------------
// Phase 2: persistent scan. Block j owns H columns [j*16, j*16+16).
// 4 waves; wave w owns B rows [w*16, w*16+16). Per step, per wave:
//   24x v_wmma_f32_16x16x32_bf16 (3 gates x K=256), then fused gate math.
// Wh B-fragments live in LDS (24 KB), packed once in exact WMMA layout.
// gi values for step t+1 are loaded during step t (in flight across barrier).
// No barrier after the last step of a chunk: the kernel boundary orders it.
// ---------------------------------------------------------------------------
__global__ void __launch_bounds__(128) gru_scan_kernel(
    const float* __restrict__ gi, const float* __restrict__ Wh,
    const float* __restrict__ bhn, const float* __restrict__ h0,
    float* __restrict__ hb0, float* __restrict__ hb1,
    unsigned* __restrict__ ctrl, float* __restrict__ out,
    int Tc, int chunk, unsigned pbase) {
  __shared__ v16bf smB[3][8][32];  // [gate][kchunk][lane]

  int tid = threadIdx.x;
  int wv = tid >> 5, lane = tid & 31, l16 = lane & 15, lh = lane >> 4;
  int j = blockIdx.x;
  unsigned* cnt = ctrl;
  unsigned* gen = ctrl + 1;

  // One-time: pack this block's 48 Wh columns into LDS as bf16 B-fragments.
  for (int f = wv; f < 24; f += 4) {
    int g = f >> 3, kk = f & 7;
    smB[g][kk][lane] = load_b_frag(Wh, GRU_3H, kk * 32, g * GRU_H + j * 16, lane);
  }

  if (chunk == 0) {
    // Block j seeds h rows [j*4, j*4+4) of hb0 from h0, then barrier (phase 1).
    for (int idx = tid; idx < 4 * GRU_H; idx += 128) {
      int rr = j * 4 + (idx >> 8);
      hb0[(size_t)rr * GRU_H + (idx & 255)] = h0[(size_t)rr * GRU_H + (idx & 255)];
    }
    grid_barrier(cnt, gen, 1u);
    __syncthreads();
  } else {
    __syncthreads();  // cover smB fill
  }

  int cH = j * 16 + l16;        // this lane's H column for gate math
  float bhn_c = bhn[cH];
  int rowA = wv * 16 + l16;     // this lane's h row for the A-fragment

  // Software pipeline: gi gate values for the current step, one step ahead.
  float pg[24];  // [0..7]=r, [8..15]=z, [16..23]=n  (index v = C/D row slot)
#pragma unroll
  for (int v = 0; v < 8; ++v) {
    int b = wv * 16 + lh * 8 + v;
    const float* gir = gi + ((size_t)b * Tc + 0) * GRU_3H;
    pg[v]      = gir[cH];
    pg[8 + v]  = gir[GRU_H + cH];
    pg[16 + v] = gir[2 * GRU_H + cH];
  }

  for (int tl = 0; tl < Tc; ++tl) {
    const float* hr = (tl & 1) ? hb1 : hb0;   // Tc is even -> live h at hb0
    float*       hw = (tl & 1) ? hb0 : hb1;

    v8f acc0 = {}, acc1 = {}, acc2 = {};
    const float* arow = hr + (size_t)rowA * GRU_H + lh * 8;
#pragma unroll
    for (int kk = 0; kk < 8; ++kk) {
      v16bf a = load_a_frag(arow + kk * 32);
      acc0 = wmma_bf16(a, smB[0][kk][lane], acc0);
      acc1 = wmma_bf16(a, smB[1][kk][lane], acc1);
      acc2 = wmma_bf16(a, smB[2][kk][lane], acc2);
    }

    // Issue next step's gi loads now; they stay in flight across the barrier.
    float ng[24];
    if (tl + 1 < Tc) {
#pragma unroll
      for (int v = 0; v < 8; ++v) {
        int b = wv * 16 + lh * 8 + v;
        const float* gir = gi + ((size_t)b * Tc + (tl + 1)) * GRU_3H;
        ng[v]      = gir[cH];
        ng[8 + v]  = gir[GRU_H + cH];
        ng[16 + v] = gir[2 * GRU_H + cH];
      }
    } else {
#pragma unroll
      for (int q = 0; q < 24; ++q) ng[q] = 0.f;
    }

    int t = chunk * Tc + tl;
#pragma unroll
    for (int v = 0; v < 8; ++v) {
      int b = wv * 16 + lh * 8 + v;  // C/D layout row
      float r = sigm(pg[v] + acc0[v]);
      float z = sigm(pg[8 + v] + acc1[v]);
      float n = tanhf(pg[16 + v] + r * (acc2[v] + bhn_c));
      float ho = hr[(size_t)b * GRU_H + cH];
      float hn = (1.f - z) * ho + z * n;
      hw[(size_t)b * GRU_H + cH] = hn;
      // ys[b, t, c] after hT block
      out[(size_t)GRU_B * GRU_H + ((size_t)b * GRU_T + t) * GRU_H + cH] = hn;
      if (t == GRU_T - 1) out[(size_t)b * GRU_H + cH] = hn;  // hT
    }

    if (tl + 1 < Tc)  // last step of the chunk: kernel boundary is the barrier
      grid_barrier(cnt, gen, pbase + (unsigned)tl + 1u);

#pragma unroll
    for (int q = 0; q < 24; ++q) pg[q] = ng[q];
  }
}

// ---------------------------------------------------------------------------
extern "C" void kernel_launch(void* const* d_in, const int* in_sizes, int n_in,
                              void* d_out, int out_size, void* d_ws, size_t ws_size,
                              hipStream_t stream) {
  const float* x   = (const float*)d_in[0];
  const float* h0  = (const float*)d_in[1];
  const float* Wi  = (const float*)d_in[2];
  const float* bi  = (const float*)d_in[3];
  const float* Wh  = (const float*)d_in[4];
  const float* bhn = (const float*)d_in[5];
  float* out = (float*)d_out;

  // Workspace: [gi chunk: B*Tc*3H f32][hb0: B*H f32][hb1: B*H f32][ctrl: 2 u32]
  const size_t fixed = 2ull * GRU_B * GRU_H * sizeof(float) + 256;
  int Tc = GRU_T;
  while (Tc > 2 && (size_t)GRU_B * Tc * GRU_3H * sizeof(float) + fixed > ws_size)
    Tc >>= 1;
  int tcsh = 31 - __builtin_clz((unsigned)Tc);

  char* p = (char*)d_ws;
  float* gi = (float*)p;              p += (size_t)GRU_B * Tc * GRU_3H * sizeof(float);
  float* hb0 = (float*)p;             p += (size_t)GRU_B * GRU_H * sizeof(float);
  float* hb1 = (float*)p;             p += (size_t)GRU_B * GRU_H * sizeof(float);
  unsigned* ctrl = (unsigned*)p;

  hipMemsetAsync(ctrl, 0, 2 * sizeof(unsigned), stream);

  int nchunks = GRU_T / Tc;
  int rowtiles = (GRU_B * Tc) / 16;
  int gemm_blocks = (rowtiles * 48) / 8;  // 8 waves (tiles) per 256-thread block
  for (int c = 0; c < nchunks; ++c) {
    // Barriers completed before chunk c's first step barrier:
    //   chunk 0: 1 (the h0-copy barrier); chunk c>0: Tc + (c-1)*(Tc-1)
    unsigned pbase = (c == 0) ? 1u : (unsigned)(Tc + (c - 1) * (Tc - 1));
    gi_gemm_kernel<<<gemm_blocks, 256, 0, stream>>>(x, Wi, bi, gi, Tc, tcsh, c * Tc);
    gru_scan_kernel<<<NBLK, 128, 0, stream>>>(gi, Wh, bhn, h0, hb0, hb1, ctrl, out,
                                              Tc, c, pbase);
  }
}
